// MoE1_90297392431445
// MI455X (gfx1250) — matrix operand order repaired and checked by reading, not compiled, not run on hardware
//
#include <hip/hip_runtime.h>
#include <hip/hip_bf16.h>

// Problem constants (B=4, S=2048 -> T=8192; D=1024; E=8; H=4096; top-2).
#define TOKENS 8192
#define DMODEL 1024
#define NEXP   8
#define HDIM   4096
#define MTILE  64          // tokens per expert block (4 m-tiles of 16)

typedef __bf16 bf16;
typedef bf16  v16bf __attribute__((ext_vector_type(16)));
typedef bf16  v4bf  __attribute__((ext_vector_type(4)));
typedef float v8f   __attribute__((ext_vector_type(8)));
typedef float v4f   __attribute__((ext_vector_type(4)));

// ---------------------------------------------------------------------------
// WMMA helpers (CDNA5 wave32, 16x16x32 bf16 -> f32)
// ---------------------------------------------------------------------------
__device__ __forceinline__ v8f wmma_bf16(v16bf a, v16bf b, v8f c) {
    return __builtin_amdgcn_wmma_f32_16x16x32_bf16(
        false, a, false, b, (short)0, c, false, false);
}

// A-matrix 16x32 bf16 lane layout (ISA 7.12.2):
//   lanes 0-15 : M=lane,    K = {0..7, 16..23}
//   lanes 16-31: M=lane-16, K = {8..15, 24..31}
__device__ __forceinline__ v16bf load_a_frag(const bf16* base, int stride, int lane) {
    const int m  = lane & 15;
    const int kb = (lane < 16) ? 0 : 8;
    const bf16* p = base + m * stride + kb;
    v16bf a;
#pragma unroll
    for (int i = 0; i < 8; ++i) a[i] = p[i];
#pragma unroll
    for (int i = 0; i < 8; ++i) a[8 + i] = p[16 + i];
    return a;
}

// B-matrix 32x16 bf16 lane layout:
//   lanes 0-15 : N=lane, K = {0..7, 16..23} ; lanes 16-31: N=lane-16, K = {8..15, 24..31}
__device__ __forceinline__ v16bf load_b_frag(const bf16* baseT, int stride, int lane) {
    const int n  = lane & 15;
    const int kb = (lane < 16) ? 0 : 8;
    const bf16* p = baseT + n * stride + kb;
    v16bf b;
#pragma unroll
    for (int i = 0; i < 8; ++i) b[i] = p[i];
#pragma unroll
    for (int i = 0; i < 8; ++i) b[8 + i] = p[16 + i];
    return b;
}

__device__ __forceinline__ float gelu_exact(float v) {
    return 0.5f * v * (1.0f + erff(v * 0.70710678118654752f));
}

// ---------------------------------------------------------------------------
// CDNA5 async copy: global -> LDS, 16B per lane, tracked by ASYNCcnt.
// LDS operand = low 32 bits of the generic shared pointer (wave-relative
// LDS byte offset per ISA 10.2 aperture layout).
// ---------------------------------------------------------------------------
__device__ __forceinline__ unsigned lds_off(const void* p) {
    return (unsigned)(size_t)p;
}
__device__ __forceinline__ void async_copy16(unsigned ldsOff, const void* g) {
    asm volatile("global_load_async_to_lds_b128 %0, %1, off"
                 :: "v"(ldsOff), "v"(g) : "memory");
}
__device__ __forceinline__ void wait_async0() {
    asm volatile("s_wait_asynccnt 0x0" ::: "memory");
}

// ---------------------------------------------------------------------------
// Stage a [ROWS x COLS] f32 global chunk into LDS transposed as [COLS][LDK]
// bf16 (fallback path when workspace is too small for bf16 weights).
// ---------------------------------------------------------------------------
template <int ROWS, int COLS, int LDK>
__device__ __forceinline__ void stage_T(bf16* dstT, const float* src,
                                        int srcStride, int tIdx,
                                        const float* pf /*nullable*/) {
    constexpr int NMT = (ROWS * COLS) / 16;
    constexpr int MTC = COLS / 4;
#pragma unroll
    for (int it = 0; it < NMT / 256; ++it) {
        const int mt = it * 256 + tIdx;
        const int r4 = (mt / MTC) * 4;
        const int c4 = (mt % MTC) * 4;
        const float* s = src + (size_t)r4 * srcStride + c4;
        const v4f r0 = *(const v4f*)(s);
        const v4f r1 = *(const v4f*)(s + srcStride);
        const v4f r2 = *(const v4f*)(s + 2 * srcStride);
        const v4f r3 = *(const v4f*)(s + 3 * srcStride);
        if (pf) __builtin_prefetch(pf + (size_t)r4 * srcStride + c4, 0, 1);
#pragma unroll
        for (int c = 0; c < 4; ++c) {
            const v4f col = {r0[c], r1[c], r2[c], r3[c]};
            *(v4bf*)(dstT + (size_t)(c4 + c) * LDK + r4) =
                __builtin_convertvector(col, v4bf);
        }
    }
}

// ---------------------------------------------------------------------------
// Kernel 0: zero output + expert counts (vectorized)
// ---------------------------------------------------------------------------
__global__ void moe_zero_kernel(float* __restrict__ out, int* __restrict__ counts,
                                size_t n4) {
    const size_t stride = (size_t)gridDim.x * blockDim.x;
    v4f* o4 = (v4f*)out;
    for (size_t i = (size_t)blockIdx.x * blockDim.x + threadIdx.x; i < n4; i += stride)
        o4[i] = v4f{};
    if (blockIdx.x == 0 && threadIdx.x < NEXP) counts[threadIdx.x] = 0;
}

// ---------------------------------------------------------------------------
// Kernel 1: router — one wave (32 lanes) per token
// ---------------------------------------------------------------------------
__global__ void moe_router_kernel(const float* __restrict__ x,
                                  const float* __restrict__ Wr,
                                  const float* __restrict__ br,
                                  int* __restrict__ topi,
                                  float* __restrict__ topw) {
    const int wv   = threadIdx.x >> 5;
    const int lane = threadIdx.x & 31;
    const int t    = blockIdx.x * 8 + wv;
    if (t >= TOKENS) return;

    float acc[NEXP] = {};
    for (int d = lane; d < DMODEL; d += 32) {
        const float xv = x[(size_t)t * DMODEL + d];
        const float* wr = Wr + (size_t)d * NEXP;
#pragma unroll
        for (int e = 0; e < NEXP; ++e) acc[e] += xv * wr[e];
    }
#pragma unroll
    for (int e = 0; e < NEXP; ++e) {
#pragma unroll
        for (int off = 16; off > 0; off >>= 1)
            acc[e] += __shfl_xor(acc[e], off, 32);
    }
    if (lane == 0) {
        float l[NEXP];
#pragma unroll
        for (int e = 0; e < NEXP; ++e) l[e] = acc[e] + br[e];
        int i0 = 0;
#pragma unroll
        for (int e = 1; e < NEXP; ++e) if (l[e] > l[i0]) i0 = e;
        int i1 = (i0 == 0) ? 1 : 0;
#pragma unroll
        for (int e = 0; e < NEXP; ++e) if (e != i0 && l[e] > l[i1]) i1 = e;
        const float e2 = expf(l[i1] - l[i0]);
        const float s  = 1.0f + e2;
        topi[2 * t]     = i0;
        topi[2 * t + 1] = i1;
        topw[2 * t]     = 1.0f / s;
        topw[2 * t + 1] = e2 / s;
    }
}

// ---------------------------------------------------------------------------
// Kernel 2: scatter tokens into per-expert lists
// ---------------------------------------------------------------------------
__global__ void moe_scatter_kernel(const int* __restrict__ topi,
                                   const float* __restrict__ topw,
                                   int* __restrict__ counts,
                                   int* __restrict__ list,
                                   float* __restrict__ wl) {
    const int t = blockIdx.x * 256 + threadIdx.x;
    if (t >= TOKENS) return;
#pragma unroll
    for (int j = 0; j < 2; ++j) {
        const int e   = topi[2 * t + j];
        const int pos = atomicAdd(&counts[e], 1);
        list[(size_t)e * TOKENS + pos] = t;
        wl[(size_t)e * TOKENS + pos]   = topw[2 * t + j];
    }
}

// ---------------------------------------------------------------------------
// Kernel 2b: convert x (f32) -> xbf (bf16)
// ---------------------------------------------------------------------------
__global__ void moe_xbf_kernel(const float* __restrict__ x, bf16* __restrict__ xbf,
                               size_t n4) {
    const size_t stride = (size_t)gridDim.x * blockDim.x;
    for (size_t i = (size_t)blockIdx.x * blockDim.x + threadIdx.x; i < n4; i += stride) {
        const v4f v = *(const v4f*)(x + i * 4);
        *(v4bf*)(xbf + i * 4) = __builtin_convertvector(v, v4bf);
    }
}

// ---------------------------------------------------------------------------
// Kernel 2c: transpose+convert expert weights.
// src: [E][R][C] f32  ->  dst: [E][C][R] bf16.  64x64 tiles via LDS.
// ---------------------------------------------------------------------------
__global__ void __launch_bounds__(256)
moe_convT_kernel(const float* __restrict__ src, bf16* __restrict__ dst,
                 int R, int C) {
    const int e  = blockIdx.z;
    const int r0 = blockIdx.y * 64;
    const int c0 = blockIdx.x * 64;
    const int t  = threadIdx.x;
    __shared__ float tile[64][65];

    const float* s = src + (size_t)e * R * C;
#pragma unroll
    for (int it = 0; it < 16; ++it) {
        const int i = it * 256 + t;
        const int r = i >> 6, c = i & 63;
        tile[r][c] = s[(size_t)(r0 + r) * C + c0 + c];
    }
    __syncthreads();

    bf16* d = dst + (size_t)e * C * R;
#pragma unroll
    for (int it = 0; it < 4; ++it) {
        const int i  = it * 256 + t;
        const int c  = i >> 4;
        const int r4 = (i & 15) * 4;
        const v4f col = {tile[r4 + 0][c], tile[r4 + 1][c],
                         tile[r4 + 2][c], tile[r4 + 3][c]};
        *(v4bf*)(d + (size_t)(c0 + c) * R + r0 + r4) =
            __builtin_convertvector(col, v4bf);
    }
}

// ---------------------------------------------------------------------------
// Kernel 3a (main path): expert FFN with pre-transposed bf16 weights and
// async global->LDS staging. Block = 256 threads (8 wave32), 64-token tile.
//   xbf : [T][D]  bf16      W1T : [E][H][D] bf16      W2T : [E][D][H] bf16
// ---------------------------------------------------------------------------
__global__ void __launch_bounds__(256)
moe_expert_bf16_kernel(const bf16* __restrict__ xbf,
                       const bf16* __restrict__ W1T,
                       const float* __restrict__ b1,
                       const bf16* __restrict__ W2T,
                       const float* __restrict__ b2,
                       const int* __restrict__ counts,
                       const int* __restrict__ list,
                       const float* __restrict__ wl,
                       float* __restrict__ out) {
    const int e    = blockIdx.y;
    const int n_e  = counts[e];
    const int base = blockIdx.x * MTILE;
    if (base >= n_e) return;

    const int tIdx = threadIdx.x;
    const int wv   = tIdx >> 5;
    const int lane = tIdx & 31;

    __shared__ bf16  Xs[MTILE][32];      //  4 KB: X chunk (A, phase 1)
    __shared__ bf16  W1sT[128][32];      //  8 KB: W1 chunk, K-contig (B, phase 1)
    __shared__ bf16  Hs[MTILE][128];     // 16 KB: gelu activations (A, phase 2)
    __shared__ bf16  W2sT[256][32];      // 16 KB: W2 quarter chunk (B, phase 2)
    __shared__ int   tid_s[MTILE];
    __shared__ float w_s[MTILE];

    if (tIdx < MTILE) {
        const int idx = base + tIdx;
        if (idx < n_e) {
            tid_s[tIdx] = list[(size_t)e * TOKENS + idx];
            w_s[tIdx]   = wl[(size_t)e * TOKENS + idx];
        } else {  // pad row: duplicate a valid token, weight 0 => adds exact 0.0
            tid_s[tIdx] = list[(size_t)e * TOKENS + base];
            w_s[tIdx]   = 0.0f;
        }
    }
    __syncthreads();

    const bf16* W1Te = W1T + (size_t)e * HDIM * DMODEL;
    const bf16* W2Te = W2T + (size_t)e * DMODEL * HDIM;
    const int seg = lane & 3;              // 16B segment within a 64B row

    // Per-thread async-copy assignments (8 bf16 = 16 bytes per lane).
    // X chunk [64][32]: 1 instr/wave; row m = wv*8 + lane/4.
    const int xm = wv * 8 + (lane >> 2);
    const bf16* xg = xbf + (size_t)tid_s[xm] * DMODEL + seg * 8;
    const unsigned xl = lds_off(&Xs[0][0]) + xm * 64 + seg * 16;
    // W1 chunk [128][32]: 2 instr/wave; rows wv*16 + i*8 + lane/4.
    const int w1r0 = wv * 16 + (lane >> 2);
    const unsigned w1l0 = lds_off(&W1sT[0][0]) + w1r0 * 64 + seg * 16;
    const unsigned w1l1 = w1l0 + 8 * 64;
    // W2 chunk [256][32]: 4 instr/wave; rows wv*32 + i*8 + lane/4.
    const int w2r0 = wv * 32 + (lane >> 2);
    const unsigned w2l0 = lds_off(&W2sT[0][0]) + w2r0 * 64 + seg * 16;

    // acc[mt][jj]: m-tile mt (rows mt*16..), n-tile nt = wv + 8*jj
    v8f acc[4][8];
#pragma unroll
    for (int mt = 0; mt < 4; ++mt)
#pragma unroll
        for (int j = 0; j < 8; ++j) acc[mt][j] = v8f{};

#pragma unroll 1
    for (int h0 = 0; h0 < HDIM; h0 += 128) {
        // ---- Phase 1: Hs[64][128] = gelu(X @ W1T[h0:h0+128, :] + b1)
        v8f hacc[4];
#pragma unroll
        for (int mt = 0; mt < 4; ++mt) hacc[mt] = v8f{};

#pragma unroll 1
        for (int d0 = 0; d0 < DMODEL; d0 += 32) {
            __syncthreads();   // prior chunk fully consumed
            async_copy16(xl, xg + d0);
            const bf16* w1g = W1Te + (size_t)h0 * DMODEL + d0 + seg * 8;
            async_copy16(w1l0, w1g + (size_t)w1r0 * DMODEL);
            async_copy16(w1l1, w1g + (size_t)(w1r0 + 8) * DMODEL);
            wait_async0();
            __syncthreads();
            const v16bf b = load_b_frag(&W1sT[wv * 16][0], 32, lane);
#pragma unroll
            for (int mt = 0; mt < 4; ++mt) {
                const v16bf a = load_a_frag(&Xs[mt * 16][0], 32, lane);
                hacc[mt] = wmma_bf16(a, b, hacc[mt]);
            }
        }
        // bias + exact gelu, write bf16 activations to LDS
        {
            const int   n   = lane & 15;
            const int   mb  = (lane < 16) ? 0 : 8;
            const float b1v = b1[(size_t)e * HDIM + h0 + wv * 16 + n];
#pragma unroll
            for (int mt = 0; mt < 4; ++mt)
#pragma unroll
                for (int r = 0; r < 8; ++r) {
                    const float v = gelu_exact(hacc[mt][r] + b1v);
                    Hs[mt * 16 + mb + r][wv * 16 + n] = (bf16)v;
                }
        }
        __syncthreads();

        // ---- Phase 2: acc += Hs @ W2T[:, h0:h0+128], 256-col quarters
#pragma unroll 1
        for (int kk0 = 0; kk0 < 128; kk0 += 32) {
            v16bf aH[4];
#pragma unroll
            for (int mt = 0; mt < 4; ++mt)
                aH[mt] = load_a_frag(&Hs[mt * 16][kk0], 128, lane);
#pragma unroll 1
            for (int q = 0; q < 4; ++q) {
                __syncthreads();
                const bf16* w2g = W2Te + (size_t)q * 256 * HDIM + h0 + kk0 + seg * 8;
#pragma unroll
                for (int i = 0; i < 4; ++i)
                    async_copy16(w2l0 + i * 8 * 64,
                                 w2g + (size_t)(w2r0 + i * 8) * HDIM);
                wait_async0();
                __syncthreads();
#pragma unroll
                for (int u = 0; u < 2; ++u) {
                    const int jj = 2 * q + u;
                    const int nt = wv + 8 * jj;
                    const int colT = nt * 16 - q * 256;
                    const v16bf bW = load_b_frag(&W2sT[colT][0], 32, lane);
#pragma unroll
                    for (int mt = 0; mt < 4; ++mt)
                        acc[mt][jj] = wmma_bf16(aH[mt], bW, acc[mt][jj]);
                }
            }
        }
    }

    // ---- Finalize: out[t] += w * (acc + b2[e])
    {
        const int n  = lane & 15;
        const int mb = (lane < 16) ? 0 : 8;
#pragma unroll
        for (int jj = 0; jj < 8; ++jj) {
            const int   col = (wv + 8 * jj) * 16 + n;
            const float b2v = b2[(size_t)e * DMODEL + col];
#pragma unroll
            for (int mt = 0; mt < 4; ++mt)
#pragma unroll
                for (int r = 0; r < 8; ++r) {
                    const int   M   = mt * 16 + mb + r;
                    const float val = w_s[M] * (acc[mt][jj][r] + b2v);
                    atomicAdd(&out[(size_t)tid_s[M] * DMODEL + col], val);
                }
        }
    }
}

// ---------------------------------------------------------------------------
// Kernel 3b (fallback): expert FFN staging f32 weights with on-the-fly
// bf16 conversion (used when workspace is too small for bf16 weights).
// ---------------------------------------------------------------------------
__global__ void __launch_bounds__(256)
moe_expert_kernel(const float* __restrict__ x,
                  const float* __restrict__ W1,
                  const float* __restrict__ b1,
                  const float* __restrict__ W2,
                  const float* __restrict__ b2,
                  const int* __restrict__ counts,
                  const int* __restrict__ list,
                  const float* __restrict__ wl,
                  float* __restrict__ out) {
    const int e    = blockIdx.y;
    const int n_e  = counts[e];
    const int base = blockIdx.x * MTILE;
    if (base >= n_e) return;

    const int tIdx = threadIdx.x;
    const int wv   = tIdx >> 5;
    const int lane = tIdx & 31;

    __shared__ bf16  Xs[MTILE][32];
    __shared__ bf16  W1sT[128][32];
    __shared__ bf16  Hs[MTILE][128];
    __shared__ bf16  W2sT[256][32];
    __shared__ int   tid_s[MTILE];
    __shared__ float w_s[MTILE];

    if (tIdx < MTILE) {
        const int idx = base + tIdx;
        if (idx < n_e) {
            tid_s[tIdx] = list[(size_t)e * TOKENS + idx];
            w_s[tIdx]   = wl[(size_t)e * TOKENS + idx];
        } else {
            tid_s[tIdx] = list[(size_t)e * TOKENS + base];
            w_s[tIdx]   = 0.0f;
        }
    }
    __syncthreads();

    const int   m0    = tIdx >> 3;
    const int   f4    = (tIdx & 7) * 4;
    const float* xrow0 = x + (size_t)tid_s[m0] * DMODEL;
    const float* xrow1 = x + (size_t)tid_s[m0 + 32] * DMODEL;

    v8f acc[4][8];
#pragma unroll
    for (int mt = 0; mt < 4; ++mt)
#pragma unroll
        for (int j = 0; j < 8; ++j) acc[mt][j] = v8f{};

    const float* W1e = W1 + (size_t)e * DMODEL * HDIM;
    const float* W2e = W2 + (size_t)e * HDIM * DMODEL;

#pragma unroll 1
    for (int h0 = 0; h0 < HDIM; h0 += 128) {
        v8f hacc[4];
#pragma unroll
        for (int mt = 0; mt < 4; ++mt) hacc[mt] = v8f{};

#pragma unroll 1
        for (int d0 = 0; d0 < DMODEL; d0 += 32) {
            __syncthreads();
            *(v4bf*)(&Xs[m0][f4]) =
                __builtin_convertvector(*(const v4f*)(xrow0 + d0 + f4), v4bf);
            *(v4bf*)(&Xs[m0 + 32][f4]) =
                __builtin_convertvector(*(const v4f*)(xrow1 + d0 + f4), v4bf);
            const float* src = W1e + (size_t)d0 * HDIM + h0;
            const float* pf  = (d0 + 32 < DMODEL) ? (src + (size_t)32 * HDIM) : nullptr;
            stage_T<32, 128, 32>(&W1sT[0][0], src, HDIM, tIdx, pf);
            __syncthreads();
            const v16bf b = load_b_frag(&W1sT[wv * 16][0], 32, lane);
#pragma unroll
            for (int mt = 0; mt < 4; ++mt) {
                const v16bf a = load_a_frag(&Xs[mt * 16][0], 32, lane);
                hacc[mt] = wmma_bf16(a, b, hacc[mt]);
            }
        }
        {
            const int   n   = lane & 15;
            const int   mb  = (lane < 16) ? 0 : 8;
            const float b1v = b1[(size_t)e * HDIM + h0 + wv * 16 + n];
#pragma unroll
            for (int mt = 0; mt < 4; ++mt)
#pragma unroll
                for (int r = 0; r < 8; ++r) {
                    const float v = gelu_exact(hacc[mt][r] + b1v);
                    Hs[mt * 16 + mb + r][wv * 16 + n] = (bf16)v;
                }
        }
        __syncthreads();

#pragma unroll 1
        for (int kk0 = 0; kk0 < 128; kk0 += 32) {
            v16bf aH[4];
#pragma unroll
            for (int mt = 0; mt < 4; ++mt)
                aH[mt] = load_a_frag(&Hs[mt * 16][kk0], 128, lane);
#pragma unroll 1
            for (int q = 0; q < 4; ++q) {
                __syncthreads();
                const float* src = W2e + (size_t)(h0 + kk0) * DMODEL + q * 256;
                stage_T<32, 256, 32>(&W2sT[0][0], src, DMODEL, tIdx, nullptr);
                __syncthreads();
#pragma unroll
                for (int u = 0; u < 2; ++u) {
                    const int jj = 2 * q + u;
                    const int nt = wv + 8 * jj;
                    const int colT = nt * 16 - q * 256;
                    const v16bf bW = load_b_frag(&W2sT[colT][0], 32, lane);
#pragma unroll
                    for (int mt = 0; mt < 4; ++mt)
                        acc[mt][jj] = wmma_bf16(aH[mt], bW, acc[mt][jj]);
                }
            }
        }
    }

    {
        const int n  = lane & 15;
        const int mb = (lane < 16) ? 0 : 8;
#pragma unroll
        for (int jj = 0; jj < 8; ++jj) {
            const int   col = (wv + 8 * jj) * 16 + n;
            const float b2v = b2[(size_t)e * DMODEL + col];
#pragma unroll
            for (int mt = 0; mt < 4; ++mt)
#pragma unroll
                for (int r = 0; r < 8; ++r) {
                    const int   M   = mt * 16 + mb + r;
                    const float val = w_s[M] * (acc[mt][jj][r] + b2v);
                    atomicAdd(&out[(size_t)tid_s[M] * DMODEL + col], val);
                }
        }
    }
}

// ---------------------------------------------------------------------------
// Host launch
// ---------------------------------------------------------------------------
extern "C" void kernel_launch(void* const* d_in, const int* in_sizes, int n_in,
                              void* d_out, int out_size, void* d_ws, size_t ws_size,
                              hipStream_t stream) {
    const float* x  = (const float*)d_in[0];
    const float* Wr = (const float*)d_in[1];
    const float* br = (const float*)d_in[2];
    const float* W1 = (const float*)d_in[3];
    const float* b1 = (const float*)d_in[4];
    const float* W2 = (const float*)d_in[5];
    const float* b2 = (const float*)d_in[6];
    float* out = (float*)d_out;

    char* ws = (char*)d_ws;
    size_t off = 0;
    int*   counts = (int*)(ws + off);  off += 256;
    int*   topi   = (int*)(ws + off);  off += (size_t)2 * TOKENS * 4;
    float* topw   = (float*)(ws + off); off += (size_t)2 * TOKENS * 4;
    int*   list   = (int*)(ws + off);  off += (size_t)NEXP * TOKENS * 4;
    float* wl     = (float*)(ws + off); off += (size_t)NEXP * TOKENS * 4;
    const size_t routingBytes = off;
    bf16* xbf = (bf16*)(ws + off);     off += (size_t)TOKENS * DMODEL * 2;
    bf16* W1T = (bf16*)(ws + off);     off += (size_t)NEXP * DMODEL * HDIM * 2;
    bf16* W2T = (bf16*)(ws + off);     off += (size_t)NEXP * DMODEL * HDIM * 2;
    const size_t needBf16 = off;

    moe_zero_kernel<<<2048, 256, 0, stream>>>(out, counts,
                                              (size_t)TOKENS * DMODEL / 4);
    moe_router_kernel<<<TOKENS / 8, 256, 0, stream>>>(x, Wr, br, topi, topw);
    moe_scatter_kernel<<<TOKENS / 256, 256, 0, stream>>>(topi, topw, counts, list, wl);

    dim3 grid(TOKENS / MTILE, NEXP);
    if (ws_size >= needBf16) {
        moe_xbf_kernel<<<2048, 256, 0, stream>>>(x, xbf,
                                                 (size_t)TOKENS * DMODEL / 4);
        dim3 g1(HDIM / 64, DMODEL / 64, NEXP);   // W1 [D][H] -> W1T [H][D]
        moe_convT_kernel<<<g1, 256, 0, stream>>>(W1, W1T, DMODEL, HDIM);
        dim3 g2(DMODEL / 64, HDIM / 64, NEXP);   // W2 [H][D] -> W2T [D][H]
        moe_convT_kernel<<<g2, 256, 0, stream>>>(W2, W2T, HDIM, DMODEL);
        moe_expert_bf16_kernel<<<grid, 256, 0, stream>>>(xbf, W1T, b1, W2T, b2,
                                                         counts, list, wl, out);
    } else {
        (void)routingBytes;
        moe_expert_kernel<<<grid, 256, 0, stream>>>(x, W1, b1, W2, b2,
                                                    counts, list, wl, out);
    }
}
